// RCG_74680891343395
// MI455X (gfx1250) — matrix operand order repaired and checked by reading, not compiled, not run on hardware
//
#include <hip/hip_runtime.h>
#include <hip/hip_bf16.h>
#include <math.h>

// ---------------- CDNA5 WMMA types ----------------
typedef __attribute__((ext_vector_type(16))) _Float16 v16h;
typedef __attribute__((ext_vector_type(8)))  float    v8f;

__device__ __forceinline__ float sigmoidf_(float x){ return 1.f/(1.f+__expf(-x)); }

#define LSEQ 16384            // 128*128
#define BL  131072            // 8*16384

// ---------------- WMMA fragment loaders (16x16x32 f16, contiguous k order shared by A and B) ----------------
// A (MxK f32 row-major -> f16):
//  lanes 0-15 : M=row0+lane,    elems 0..15 = K k0..k0+15
//  lanes 16-31: M=row0+lane-16, elems 0..15 = K k0+16..k0+31
__device__ __forceinline__ v16h load_a_frag(const float* __restrict__ A, int lda,
                                            int row0, int k0, int lane){
  int m  = row0 + (lane & 15);
  int kb = k0 + ((lane >> 4) << 4);
  const float* p = A + (size_t)m * lda + kb;
  v16h a;
#pragma unroll
  for (int j = 0; j < 8; ++j) a[j]   = (_Float16)p[j];
#pragma unroll
  for (int j = 0; j < 8; ++j) a[8+j] = (_Float16)p[8+j];
  return a;
}

// B (KxN) from W [N][K] row-major (X @ W^T):
//  lanes 0-15: K k0..k0+15 ; lanes 16-31: K k0+16..k0+31 ; N = col0 + (lane&15)
__device__ __forceinline__ v16h load_b_frag(const float* __restrict__ W, int ldw,
                                            int col0, int k0, int lane, int nmax){
  int n  = col0 + (lane & 15);
  int kb = k0 + ((lane >> 4) << 4);
  v16h bf;
  if (n < nmax){
    const float* p = W + (size_t)n * ldw + kb;
#pragma unroll
    for (int j = 0; j < 16; ++j) bf[j] = (_Float16)p[j];
  } else {
#pragma unroll
    for (int j = 0; j < 16; ++j) bf[j] = (_Float16)0.f;
  }
  return bf;
}

// A-fragment from an LDS-staged 16x32 tile (row stride 40 halfs, padded)
__device__ __forceinline__ v16h load_a_frag_lds(const _Float16 (*As)[40], int lane){
  int mrow = lane & 15;
  int kb   = (lane >> 4) << 4;
  v16h a;
#pragma unroll
  for (int j = 0; j < 8; ++j) a[j]   = As[mrow][kb + j];
#pragma unroll
  for (int j = 0; j < 8; ++j) a[8+j] = As[mrow][kb + 8 + j];
  return a;
}

// ---------------- K1: bilinear(edge 32->64, align-corners) + (1-sigmoid(pre))*f -> x1 [B,128,64,64]
__global__ void k1_prep(const float* __restrict__ pre, const float* __restrict__ edge,
                        const float* __restrict__ f, float* __restrict__ x1){
  int idx = blockIdx.x * blockDim.x + threadIdx.x;       // B*128*4096
  int p = idx & 4095;
  int c = (idx >> 12) & 127;
  int b = idx >> 19;
  int y = p >> 6, x = p & 63;
  float v;
  if (c < 64){
    const float s = 31.f / 63.f;
    float ys = y * s, xs = x * s;
    int y0 = (int)ys, x0 = (int)xs;
    int y1 = min(y0 + 1, 31), x1i = min(x0 + 1, 31);
    float fy = ys - y0, fx = xs - x0;
    const float* e = edge + ((size_t)(b * 64 + c)) * 1024;
    float t00 = e[y0*32 + x0],  t01 = e[y0*32 + x1i];
    float t10 = e[y1*32 + x0],  t11 = e[y1*32 + x1i];
    float top = t00 * (1.f - fx) + t01 * fx;
    float bot = t10 * (1.f - fx) + t11 * fx;
    v = top * (1.f - fy) + bot * fy;
  } else {
    float att = sigmoidf_(pre[(size_t)b * 4096 + p]);
    v = (1.f - att) * f[((size_t)(b * 64 + (c - 64))) * 4096 + p];
  }
  x1[idx] = v;
}

// ---------------- K2: 3x3 conv (128->64, pad1) + BN + ReLU via WMMA implicit GEMM
// M = 4096 pixels/batch, N = 64, K = 128*9 = 1152; weights: B[k][n] = w[n*1152+k]
// grid: 8*256 blocks (batch, m-tile), 128 threads (4 waves = n-tiles)
__global__ void k2_conv1(const float* __restrict__ x1, const float* __restrict__ w,
                         const float* __restrict__ bconv, const float* __restrict__ bg,
                         const float* __restrict__ bb, const float* __restrict__ bm,
                         const float* __restrict__ bv, float* __restrict__ x2){
  __shared__ _Float16 As[16][40];
  int b    = blockIdx.x >> 8;
  int m0   = (blockIdx.x & 255) << 4;
  int lane = threadIdx.x & 31;
  int col0 = (threadIdx.x >> 5) << 4;
  const float* xb = x1 + (size_t)b * 128 * 4096;
  v8f acc = {};
#pragma unroll 1
  for (int k0 = 0; k0 < 1152; k0 += 32){
    // cooperative im2col gather into LDS: 512 elems / 128 threads
#pragma unroll
    for (int e = threadIdx.x; e < 512; e += 128){
      int r  = e >> 5, kc = e & 31;
      int kk = k0 + kc;
      int ic  = kk / 9;
      int rem = kk - ic * 9;
      int ky  = rem / 3;
      int kx  = rem - ky * 3;
      int p = m0 + r;
      int y = (p >> 6) + ky - 1;
      int x = (p & 63) + kx - 1;
      float v = 0.f;
      if ((unsigned)y < 64u && (unsigned)x < 64u)
        v = xb[(size_t)ic * 4096 + y*64 + x];
      As[r][kc] = (_Float16)v;
    }
    __syncthreads();
    v16h a  = load_a_frag_lds(As, lane);
    v16h bf = load_b_frag(w, 1152, col0, k0, lane, 64);
    acc = __builtin_amdgcn_wmma_f32_16x16x32_f16(false, a, false, bf, (short)0, acc, false, false);
    __syncthreads();
  }
  int n     = col0 + (lane & 15);
  int rbase = m0 + ((lane >> 4) << 3);
  float scale = bg[n] * rsqrtf(bv[n] + 1e-5f);
  float shift = (bconv[n] - bm[n]) * scale + bb[n];
  float* dst = x2 + ((size_t)(b * 64 + n)) * 4096;
#pragma unroll
  for (int r = 0; r < 8; ++r){
    float v = acc[r] * scale + shift;
    dst[rbase + r] = v > 0.f ? v : 0.f;
  }
}

// ---------------- K3: conv-transpose k4 s2 p1 (64->64) via WMMA implicit GEMM -> x0 NHWC [B,L,64]
// M = 16384 pixels/batch, N = 64, K = 64*16 = 1024 zero-padded im2col (odd/oob taps = 0)
// B[k][n] = up_w[ic*1024 + n*16 + kyx] (kb 16-aligned -> contiguous 16-elem read)
__global__ void k3_upconv(const float* __restrict__ x2, const float* __restrict__ w,
                          const float* __restrict__ bias, float* __restrict__ x0){
  __shared__ _Float16 As[16][40];
  int b    = blockIdx.x >> 10;
  int m0   = (blockIdx.x & 1023) << 4;
  int lane = threadIdx.x & 31;
  int col0 = (threadIdx.x >> 5) << 4;
  const float* xb = x2 + (size_t)b * 64 * 4096;
  v8f acc = {};
#pragma unroll 1
  for (int k0 = 0; k0 < 1024; k0 += 32){
#pragma unroll
    for (int e = threadIdx.x; e < 512; e += 128){
      int r  = e >> 5, kc = e & 31;
      int kk = k0 + kc;
      int ic = kk >> 4;
      int ky = (kk >> 2) & 3;
      int kx = kk & 3;
      int p  = m0 + r;
      int ty = (p >> 7) + ky - 2;
      int tx = (p & 127) + kx - 2;
      float v = 0.f;
      if (ty >= 0 && !(ty & 1) && (ty >> 1) < 64 &&
          tx >= 0 && !(tx & 1) && (tx >> 1) < 64)
        v = xb[(size_t)ic * 4096 + (ty >> 1)*64 + (tx >> 1)];
      As[r][kc] = (_Float16)v;
    }
    __syncthreads();
    v16h a = load_a_frag_lds(As, lane);
    // custom B gather: k = ic*16 + kyx -> up_w[ic*1024 + n*16 + kyx]
    int n  = col0 + (lane & 15);
    int kb = k0 + ((lane >> 4) << 4);
    const float* bp = w + (size_t)(kb >> 4) * 1024 + (size_t)n * 16;  // kyx = j
    v16h bf;
#pragma unroll
    for (int j = 0; j < 16; ++j) bf[j] = (_Float16)bp[j];
    acc = __builtin_amdgcn_wmma_f32_16x16x32_f16(false, a, false, bf, (short)0, acc, false, false);
    __syncthreads();
  }
  int n     = col0 + (lane & 15);
  int rbase = m0 + ((lane >> 4) << 3);
  float bs  = bias[n];
  float* dst = x0 + (size_t)b * LSEQ * 64;
#pragma unroll
  for (int r = 0; r < 8; ++r)
    dst[(size_t)(rbase + r) * 64 + n] = acc[r] + bs;
}

// ---------------- K4: in_proj WMMA GEMM: [BL,64] @ [256,64]^T -> xin[BL,128], z[BL,128]
__global__ void k4_inproj(const float* __restrict__ X, const float* __restrict__ W,
                          float* __restrict__ xin, float* __restrict__ zout){
  int lane = threadIdx.x & 31;
  int col0 = (threadIdx.x >> 5) << 4;   // 16 waves -> n-tiles
  int row0 = blockIdx.x << 4;
  v8f acc = {};
#pragma unroll
  for (int k0 = 0; k0 < 64; k0 += 32){
    v16h a = load_a_frag(X, 64, row0, k0, lane);
    v16h b = load_b_frag(W, 64, col0, k0, lane, 256);
    acc = __builtin_amdgcn_wmma_f32_16x16x32_f16(false, a, false, b, (short)0, acc, false, false);
  }
  int col   = col0 + (lane & 15);
  int rbase = row0 + ((lane >> 4) << 3);
#pragma unroll
  for (int r = 0; r < 8; ++r){
    size_t row = (size_t)(rbase + r);
    if (col < 128) xin[row * 128 + col] = acc[r];
    else           zout[row * 128 + (col - 128)] = acc[r];
  }
}

// ---------------- K5: depthwise causal conv1d k=4 + bias + SiLU -> xs [B,L,128]
__global__ void k5_conv1d(const float* __restrict__ xin, const float* __restrict__ w,
                          const float* __restrict__ bias, float* __restrict__ xs){
  int idx = blockIdx.x * blockDim.x + threadIdx.x;       // B*L*128, d fastest
  int d = idx & 127;
  int l = (idx >> 7) & 16383;
  int b = idx >> 21;
  const float* base = xin + (size_t)b * LSEQ * 128 + d;
  float acc = bias[d];
#pragma unroll
  for (int t = 0; t < 4; ++t){
    int ll = l - 3 + t;
    if (ll >= 0) acc += base[(size_t)ll * 128] * w[d*4 + t];
  }
  xs[idx] = acc * sigmoidf_(acc);
}

// ---------------- K6: x_proj WMMA GEMM: [BL,128] @ [36,128]^T -> xdbl [BL,36]
__global__ void k6_xproj(const float* __restrict__ X, const float* __restrict__ W,
                         float* __restrict__ xdbl){
  int lane = threadIdx.x & 31;
  int col0 = (threadIdx.x >> 5) << 4;   // 3 waves (N padded 36->48)
  int row0 = blockIdx.x << 4;
  v8f acc = {};
#pragma unroll
  for (int k0 = 0; k0 < 128; k0 += 32){
    v16h a = load_a_frag(X, 128, row0, k0, lane);
    v16h b = load_b_frag(W, 128, col0, k0, lane, 36);
    acc = __builtin_amdgcn_wmma_f32_16x16x32_f16(false, a, false, b, (short)0, acc, false, false);
  }
  int col   = col0 + (lane & 15);
  int rbase = row0 + ((lane >> 4) << 3);
  if (col < 36){
#pragma unroll
    for (int r = 0; r < 8; ++r)
      xdbl[(size_t)(rbase + r) * 36 + col] = acc[r];
  }
}

// ---------------- K7: selective scan, fused dt_proj+softplus and SiLU(z) gating
__global__ void k7_scan(const float* __restrict__ xs, const float* __restrict__ xdbl,
                        const float* __restrict__ z, const float* __restrict__ dtw,
                        const float* __restrict__ dtb, const float* __restrict__ A_log,
                        const float* __restrict__ Dvec, float* __restrict__ y){
  int b = blockIdx.x >> 1;
  int d = ((blockIdx.x & 1) << 6) + (threadIdx.x >> 4);
  int n = threadIdx.x & 15;
  float w0 = dtw[d*4+0], w1 = dtw[d*4+1], w2 = dtw[d*4+2], w3 = dtw[d*4+3];
  float bias = dtb[d];
  float a    = -__expf(A_log[d*16 + n]);
  float dval = Dvec[d];
  const float* rp = xdbl + (size_t)b * LSEQ * 36;
  const float* xp = xs   + (size_t)b * LSEQ * 128 + d;
  const float* zp = z    + (size_t)b * LSEQ * 128 + d;
  float*       yp = y    + (size_t)b * LSEQ * 128 + d;
  float h = 0.f;
#pragma unroll 1
  for (int l = 0; l < LSEQ; ++l){
    float dtr = rp[0]*w0 + rp[1]*w1 + rp[2]*w2 + rp[3]*w3 + bias;
    float dt  = (dtr > 20.f) ? dtr : log1pf(__expf(dtr));   // softplus
    float Bm  = rp[4 + n];
    float Cm  = rp[20 + n];
    float x   = *xp;
    h = __expf(dt * a) * h + (dt * x) * Bm;
    float c = h * Cm;
    c += __shfl_xor(c, 1, 32);
    c += __shfl_xor(c, 2, 32);
    c += __shfl_xor(c, 4, 32);
    c += __shfl_xor(c, 8, 32);
    if (n == 0){
      float zz = *zp;
      *yp = (c + x * dval) * (zz * sigmoidf_(zz));
    }
    rp += 36; xp += 128; zp += 128; yp += 128;
  }
}

// ---------------- K8: out_proj WMMA GEMM: [BL,128] @ [64,128]^T -> out [BL,64]
__global__ void k8_outproj(const float* __restrict__ X, const float* __restrict__ W,
                           float* __restrict__ out){
  int lane = threadIdx.x & 31;
  int col0 = (threadIdx.x >> 5) << 4;   // 4 waves
  int row0 = blockIdx.x << 4;
  v8f acc = {};
#pragma unroll
  for (int k0 = 0; k0 < 128; k0 += 32){
    v16h a = load_a_frag(X, 128, row0, k0, lane);
    v16h b = load_b_frag(W, 128, col0, k0, lane, 64);
    acc = __builtin_amdgcn_wmma_f32_16x16x32_f16(false, a, false, b, (short)0, acc, false, false);
  }
  int col   = col0 + (lane & 15);
  int rbase = row0 + ((lane >> 4) << 3);
#pragma unroll
  for (int r = 0; r < 8; ++r)
    out[(size_t)(rbase + r) * 64 + col] = acc[r];
}

// ---------------- K9: x3 = sigmoid(1x1 conv of x2) -> [B,4096]
__global__ void k9_mlp(const float* __restrict__ x2, const float* __restrict__ w,
                       const float* __restrict__ bias, float* __restrict__ x3){
  int idx = blockIdx.x * blockDim.x + threadIdx.x;       // B*4096
  int p = idx & 4095, b = idx >> 12;
  float acc = bias[0];
  for (int c = 0; c < 64; ++c)
    acc += x2[((size_t)(b * 64 + c)) * 4096 + p] * w[c];
  x3[idx] = sigmoidf_(acc);
}

// ---------------- K10: down-conv k4 s2 p1 via WMMA implicit GEMM + gating + residual
// M = 4096 pixels/batch, N = 64, K = 64*16 = 1024; B[k][n] = down_w[n*1024 + k]
__global__ void k10_final(const float* __restrict__ outn, const float* __restrict__ w,
                          const float* __restrict__ bias, const float* __restrict__ x3,
                          const float* __restrict__ x2, const float* __restrict__ f,
                          float* __restrict__ res){
  __shared__ _Float16 As[16][40];
  int b    = blockIdx.x >> 8;
  int m0   = (blockIdx.x & 255) << 4;
  int lane = threadIdx.x & 31;
  int col0 = (threadIdx.x >> 5) << 4;
  const float* ob = outn + (size_t)b * LSEQ * 64;
  v8f acc = {};
#pragma unroll 1
  for (int k0 = 0; k0 < 1024; k0 += 32){
#pragma unroll
    for (int e = threadIdx.x; e < 512; e += 128){
      int r  = e >> 5, kc = e & 31;
      int kk = k0 + kc;
      int ic = kk >> 4;
      int ky = (kk >> 2) & 3;
      int kx = kk & 3;
      int p  = m0 + r;
      int yy = ((p >> 6) << 1) + ky - 1;
      int xx = ((p & 63) << 1) + kx - 1;
      float v = 0.f;
      if ((unsigned)yy < 128u && (unsigned)xx < 128u)
        v = ob[((size_t)(yy*128 + xx)) * 64 + ic];
      As[r][kc] = (_Float16)v;
    }
    __syncthreads();
    v16h a  = load_a_frag_lds(As, lane);
    v16h bf = load_b_frag(w, 1024, col0, k0, lane, 64);
    acc = __builtin_amdgcn_wmma_f32_16x16x32_f16(false, a, false, bf, (short)0, acc, false, false);
    __syncthreads();
  }
  int n     = col0 + (lane & 15);
  int rbase = m0 + ((lane >> 4) << 3);
  float bn_ = bias[n];
  const float* x2p = x2 + ((size_t)(b * 64 + n)) * 4096;
  const float* fp  = f  + ((size_t)(b * 64 + n)) * 4096;
  const float* x3p = x3 + (size_t)b * 4096;
  float* rp = res + ((size_t)(b * 64 + n)) * 4096;
#pragma unroll
  for (int r = 0; r < 8; ++r){
    int p = rbase + r;
    rp[p] = (acc[r] + bn_) * x3p[p] * x2p[p] + fp[p];
  }
}

// ---------------- launch ----------------
extern "C" void kernel_launch(void* const* d_in, const int* in_sizes, int n_in,
                              void* d_out, int out_size, void* d_ws, size_t ws_size,
                              hipStream_t stream) {
  (void)in_sizes; (void)n_in; (void)out_size; (void)ws_size;
  const float* pre      = (const float*)d_in[0];
  const float* edge     = (const float*)d_in[1];
  const float* f        = (const float*)d_in[2];
  const float* conv1_w  = (const float*)d_in[3];
  const float* conv1_b  = (const float*)d_in[4];
  const float* bn_g     = (const float*)d_in[5];
  const float* bn_b     = (const float*)d_in[6];
  const float* bn_m     = (const float*)d_in[7];
  const float* bn_v     = (const float*)d_in[8];
  const float* up_w     = (const float*)d_in[9];
  const float* up_b     = (const float*)d_in[10];
  const float* down_w   = (const float*)d_in[11];
  const float* down_b   = (const float*)d_in[12];
  const float* in_proj  = (const float*)d_in[13];
  const float* c1d_w    = (const float*)d_in[14];
  const float* c1d_b    = (const float*)d_in[15];
  const float* x_proj   = (const float*)d_in[16];
  const float* dt_w     = (const float*)d_in[17];
  const float* dt_b     = (const float*)d_in[18];
  const float* A_log    = (const float*)d_in[19];
  const float* Dvec     = (const float*)d_in[20];
  const float* out_proj = (const float*)d_in[21];
  const float* mlp_w    = (const float*)d_in[22];
  const float* mlp_b    = (const float*)d_in[23];
  float* res = (float*)d_out;
  float* ws  = (float*)d_ws;

  // workspace layout (floats), with reuse: x0 region -> out, xin region -> y
  float* x1   = ws;                         // 4,194,304
  float* x2   = ws + 4194304;               // 2,097,152
  float* x0   = ws + 6291456;               // 8,388,608  (NHWC; reused as `out`)
  float* xin  = ws + 14680064;              // 16,777,216 (reused as `y`)
  float* zbuf = ws + 31457280;              // 16,777,216
  float* xs   = ws + 48234496;              // 16,777,216
  float* xdbl = ws + 65011712;              // 4,718,592
  float* x3   = ws + 69730304;              // 32,768

  k1_prep  <<<16384, 256, 0, stream>>>(pre, edge, f, x1);
  k2_conv1 <<< 2048, 128, 0, stream>>>(x1, conv1_w, conv1_b, bn_g, bn_b, bn_m, bn_v, x2); // WMMA
  k3_upconv<<< 8192, 128, 0, stream>>>(x2, up_w, up_b, x0);                               // WMMA
  k4_inproj<<< 8192, 512, 0, stream>>>(x0, in_proj, xin, zbuf);                           // WMMA
  k5_conv1d<<<65536, 256, 0, stream>>>(xin, c1d_w, c1d_b, xs);
  k6_xproj <<< 8192,  96, 0, stream>>>(xs, x_proj, xdbl);                                 // WMMA
  float* y = xin;   // x_in dead after conv1d
  k7_scan  <<<   16,1024, 0, stream>>>(xs, xdbl, zbuf, dt_w, dt_b, A_log, Dvec, y);
  float* out = x0;  // x0 dead after in_proj
  k8_outproj<<<8192, 128, 0, stream>>>(y, out_proj, out);                                 // WMMA
  k9_mlp   <<<  128, 256, 0, stream>>>(x2, mlp_w, mlp_b, x3);
  k10_final<<< 2048, 128, 0, stream>>>(out, down_w, down_b, x3, x2, f, res);              // WMMA
}
